// AISA_8117488189753
// MI455X (gfx1250) — matrix operand order repaired and checked by reading, not compiled, not run on hardware
//
#include <hip/hip_runtime.h>
#include <hip/hip_bf16.h>

typedef __bf16 bf16_t;
typedef __attribute__((ext_vector_type(16))) __bf16 v16bf;
typedef __attribute__((ext_vector_type(8)))  float  v8f;
typedef __attribute__((ext_vector_type(4)))  unsigned int v4u;
typedef __attribute__((ext_vector_type(8)))  unsigned int v8u;

// Problem constants
#define BATCH 2
#define SEQ   2048
#define EMB   1024
#define NHEAD 16
#define HDIM  64

__device__ __forceinline__ v16bf pack_v16bf(uint4 lo, uint4 hi) {
    union { uint4 q[2]; v16bf v; } u;
    u.q[0] = lo; u.q[1] = hi;
    return u.v;
}

// A-fragment (16x32, 16-bit): lane = M row; half h holds K = {8h..8h+7} U {16+8h..16+8h+7}.
// `base` must already point at rowptr + 8*half.
__device__ __forceinline__ v16bf load_afrag(const bf16_t* base) {
    uint4 lo = *(const uint4*)(base);
    uint4 hi = *(const uint4*)(base + 16);
    return pack_v16bf(lo, hi);
}

// B-fragment (32x16, 16-bit): lane = N col; half h holds K = {8h..8h+7} U {16+8h..16+8h+7}.
// `base` must already point at colptr + 8*half (memory is [N][K]).
__device__ __forceinline__ v16bf load_bfrag(const bf16_t* base) {
    uint4 lo = *(const uint4*)(base);
    uint4 hi = *(const uint4*)(base + 16);
    return pack_v16bf(lo, hi);
}

__device__ __forceinline__ v8f wmma_bf16(v16bf a, v16bf b, v8f c) {
    return __builtin_amdgcn_wmma_f32_16x16x32_bf16(false, a, false, b, (short)0, c, false, false);
}

__device__ __forceinline__ unsigned int pack2bf(float a, float b) {
    union { __bf16 h[2]; unsigned int u; } x;
    x.h[0] = (__bf16)a; x.h[1] = (__bf16)b;
    return x.u;
}

// ---------------------------------------------------------------------------
// Tensor Data Mover: 2D tile (bf16) global -> LDS, with LDS padding of
// 4 DWORDs every 32 DWORDs (row 64 bf16 -> LDS stride 72 bf16).
// D# group0/group1 per CDNA5 ISA 8.3/8.4. Issued wave-uniform (EXEC ignored).
// ---------------------------------------------------------------------------
__device__ __forceinline__ void tdm_load_2d_bf16(unsigned lds_off, const void* gptr,
                                                 unsigned tile_d0, unsigned tile_d1,
                                                 unsigned long long stride0_elems)
{
    unsigned long long ga = (unsigned long long)(size_t)gptr;
    union { unsigned u[4]; v4u v; } g0;
    g0.u[0] = 1u;                                        // count=1, user descriptor
    g0.u[1] = lds_off;                                   // lds_addr (bytes)
    g0.u[2] = (unsigned)ga;                              // global_addr[31:0]
    g0.u[3] = (unsigned)((ga >> 32) & 0x01ffffffu)       // global_addr[56:32]
            | (2u << 30);                                // type = 2 ("image")
    union { unsigned u[8]; v8u v; } g1;
    // workgroup_mask=0 | data_size=1 (2B) | pad_enable | pad_interval=4 (32 DW)
    // | pad_amount=3 (4 DW)
    g1.u[0] = (1u << 16) | (1u << 20) | (4u << 22) | (3u << 25);
    g1.u[1] = (tile_d0 & 0xffffu) << 16;                 // tensor_dim0[15:0]
    g1.u[2] = ((tile_d0 >> 16) & 0xffffu)                // tensor_dim0[31:16]
            | ((tile_d1 & 0xffffu) << 16);               // tensor_dim1[15:0]
    g1.u[3] = ((tile_d1 >> 16) & 0xffffu)                // tensor_dim1[31:16]
            | ((tile_d0 & 0xffffu) << 16);               // tile_dim0
    g1.u[4] = (tile_d1 & 0xffffu);                       // tile_dim1 (tile_dim2=0)
    g1.u[5] = (unsigned)stride0_elems;                   // tensor_dim0_stride[31:0]
    g1.u[6] = (unsigned)((stride0_elems >> 32) & 0xffffull); // stride[47:32]
    g1.u[7] = 0u;                                        // tensor_dim1_stride=0 (2D)
    asm volatile("tensor_load_to_lds %0, %1" :: "s"(g0.v), "s"(g1.v) : "memory");
}

// ---------------------------------------------------------------------------
// Kernel 1: qkv = x @ Wqkv  (M=4096, K=1024, N=3072), fp32 in, bf16 out,
// scattered into q[b,h,s,d], k[b,h,s,d], vT[b,h,d,s].
// Block 256 thr (8 waves, 4x2), tile 128x128, BK=64, double-buffered LDS.
// ---------------------------------------------------------------------------
__global__ __launch_bounds__(256)
void aisa_qkv_kernel(const float* __restrict__ x, const float* __restrict__ Wqkv,
                     bf16_t* __restrict__ qb, bf16_t* __restrict__ kb,
                     bf16_t* __restrict__ vT)
{
    __shared__ __align__(16) bf16_t sA[2][128 * 72];   // [m][k], padded stride
    __shared__ __align__(16) bf16_t sB[2][128 * 72];   // [n][k] transposed, padded

    const int tid  = threadIdx.x;
    const int lane = tid & 31;
    const int wid  = tid >> 5;
    const int half = lane >> 4;
    const int l16  = lane & 15;
    const int n0   = blockIdx.x * 128;   // over 3072
    const int m0   = blockIdx.y * 128;   // over 4096
    const int waveM = wid >> 1;          // 0..3
    const int waveN = wid & 1;           // 0..1

    const v8f vz = {0.f,0.f,0.f,0.f,0.f,0.f,0.f,0.f};
    v8f acc[2][4];
#pragma unroll
    for (int mt = 0; mt < 2; ++mt)
#pragma unroll
        for (int nt = 0; nt < 4; ++nt) acc[mt][nt] = vz;

    float4 pa[8], pb[8];
    auto fetch = [&](int k0) {
#pragma unroll
        for (int it = 0; it < 8; ++it) {
            int idx = tid + it * 256;
            pa[it] = *(const float4*)(x + (size_t)(m0 + (idx >> 4)) * EMB
                                      + k0 + ((idx & 15) << 2));
            pb[it] = *(const float4*)(Wqkv + (size_t)(k0 + (idx >> 5)) * 3072
                                      + n0 + ((idx & 31) << 2));
        }
    };
    auto stage = [&](int buf) {
#pragma unroll
        for (int it = 0; it < 8; ++it) {
            int idx = tid + it * 256;
            int r  = idx >> 4, c4 = (idx & 15) << 2;
            uint2 p; p.x = pack2bf(pa[it].x, pa[it].y); p.y = pack2bf(pa[it].z, pa[it].w);
            *(uint2*)&sA[buf][r * 72 + c4] = p;
            int rb = idx >> 5, cb4 = (idx & 31) << 2;
            sB[buf][(cb4 + 0) * 72 + rb] = (bf16_t)pb[it].x;
            sB[buf][(cb4 + 1) * 72 + rb] = (bf16_t)pb[it].y;
            sB[buf][(cb4 + 2) * 72 + rb] = (bf16_t)pb[it].z;
            sB[buf][(cb4 + 3) * 72 + rb] = (bf16_t)pb[it].w;
        }
    };

    fetch(0);
    stage(0);
    __syncthreads();

    for (int k0 = 0; k0 < EMB; k0 += 64) {
        const int buf = (k0 >> 6) & 1;
        const bool more = (k0 + 64) < EMB;
        if (more) fetch(k0 + 64);         // prefetch next tile into VGPRs

#pragma unroll
        for (int kk = 0; kk < 64; kk += 32) {
            v16bf aF[2], bF[4];
#pragma unroll
            for (int mt = 0; mt < 2; ++mt)
                aF[mt] = load_afrag(&sA[buf][(waveM * 32 + mt * 16 + l16) * 72 + kk + 8 * half]);
#pragma unroll
            for (int nt = 0; nt < 4; ++nt)
                bF[nt] = load_bfrag(&sB[buf][(waveN * 64 + nt * 16 + l16) * 72 + kk + 8 * half]);
#pragma unroll
            for (int mt = 0; mt < 2; ++mt)
#pragma unroll
                for (int nt = 0; nt < 4; ++nt)
                    acc[mt][nt] = wmma_bf16(aF[mt], bF[nt], acc[mt][nt]);
        }

        if (more) stage(buf ^ 1);         // fill the other buffer during/after compute
        __syncthreads();
    }

    // Epilogue: scatter into q / k / vT (bf16)
#pragma unroll
    for (int mt = 0; mt < 2; ++mt)
#pragma unroll
        for (int nt = 0; nt < 4; ++nt)
#pragma unroll
            for (int i = 0; i < 8; ++i) {
                int gr = m0 + waveM * 32 + mt * 16 + i + 8 * half;   // 0..4095
                int gc = n0 + waveN * 64 + nt * 16 + l16;            // 0..3071
                int b  = gr >> 11, s = gr & 2047;
                int which = gc >> 10, e = gc & 1023;
                int h = e >> 6, d = e & 63;
                int bh = b * NHEAD + h;
                bf16_t val = (bf16_t)acc[mt][nt][i];
                if (which == 0)      qb[((size_t)bh * SEQ + s) * HDIM + d] = val;
                else if (which == 1) kb[((size_t)bh * SEQ + s) * HDIM + d] = val;
                else                 vT[((size_t)bh * HDIM + d) * SEQ + s] = val;
            }
}

// ---------------------------------------------------------------------------
// Kernel 2: fused attention, TDM-staged K/V double buffering.
// One WG per (b,h, 128 q-rows); wave owns 16 q rows. Flash-style online
// softmax accumulator + separate un-normalized leaky-ReLU accumulator.
// ---------------------------------------------------------------------------
__global__ __launch_bounds__(256)
void aisa_attn_kernel(const bf16_t* __restrict__ qb, const bf16_t* __restrict__ kb,
                      const bf16_t* __restrict__ vT, const float* __restrict__ W1,
                      const float* __restrict__ W2, bf16_t* __restrict__ aout)
{
    __shared__ __align__(16) bf16_t sK[2][64 * 72];  // TDM-padded: 64 rows, stride 72
    __shared__ __align__(16) bf16_t sV[2][64 * 72];
    __shared__ __align__(16) bf16_t sP[8][16 * 64];  // wave-private P tiles
    __shared__ __align__(16) bf16_t sR[8][16 * 64];  // wave-private R tiles

    const int tid  = threadIdx.x;
    const int lane = tid & 31;
    const int wid  = tid >> 5;
    const int half = lane >> 4;
    const int l16  = lane & 15;

    const int bh   = blockIdx.x >> 4;    // 0..31  (b*16+h)
    const int qblk = blockIdx.x & 15;    // 0..15  (128-row q block)
    const int h    = bh & (NHEAD - 1);
    const int b    = bh >> 4;
    const float w1 = W1[h];
    const float w2 = W2[h];
    const float scale = 0.125f;          // 1/sqrt(64)

    const bf16_t* Qbase = qb + (size_t)bh * SEQ * HDIM;
    const bf16_t* Kbase = kb + (size_t)bh * SEQ * HDIM;
    const bf16_t* Vbase = vT + (size_t)bh * HDIM * SEQ;

    const int qr0 = qblk * 128 + wid * 16;

    // Q A-fragments stay resident for the entire S loop.
    v16bf aQ[2];
#pragma unroll
    for (int kkq = 0; kkq < 2; ++kkq)
        aQ[kkq] = load_afrag(Qbase + (size_t)(qr0 + l16) * HDIM + kkq * 32 + 8 * half);

    const v8f vz = {0.f,0.f,0.f,0.f,0.f,0.f,0.f,0.f};
    v8f accE[4], accL[4];
#pragma unroll
    for (int t = 0; t < 4; ++t) { accE[t] = vz; accL[t] = vz; }
    float mrun[8], lrun[8];
#pragma unroll
    for (int i = 0; i < 8; ++i) { mrun[i] = -1e30f; lrun[i] = 0.f; }

    bf16_t* pw = &sP[wid][0];
    bf16_t* rw = &sR[wid][0];

    // Prologue: wave 0 DMAs K/V block 0 into buffer 0 via the TDM.
    if (wid == 0) {
        tdm_load_2d_bf16((unsigned)(size_t)&sK[0][0], Kbase, 64, 64, (unsigned long long)HDIM);
        tdm_load_2d_bf16((unsigned)(size_t)&sV[0][0], Vbase, 64, 64, (unsigned long long)SEQ);
        __builtin_amdgcn_s_wait_tensorcnt(0);
    }
    __syncthreads();

    for (int j = 0; j < SEQ / 64; ++j) {
        const int buf = j & 1;
        const int kc0 = j * 64;

        // Async prefetch of the next K/V block into the other buffer.
        if (wid == 0 && (j + 1) < SEQ / 64) {
            const int kn = (j + 1) * 64;
            tdm_load_2d_bf16((unsigned)(size_t)&sK[buf ^ 1][0],
                             Kbase + (size_t)kn * HDIM, 64, 64, (unsigned long long)HDIM);
            tdm_load_2d_bf16((unsigned)(size_t)&sV[buf ^ 1][0],
                             Vbase + kn, 64, 64, (unsigned long long)SEQ);
        }

        // S = (Q @ K^T) * scale   (4 n-tiles x K=64) from LDS K tile
        v8f S[4];
#pragma unroll
        for (int t = 0; t < 4; ++t) S[t] = vz;
#pragma unroll
        for (int kk = 0; kk < 2; ++kk) {
#pragma unroll
            for (int nt = 0; nt < 4; ++nt) {
                v16bf bK = load_bfrag(&sK[buf][(nt * 16 + l16) * 72 + kk * 32 + 8 * half]);
                S[nt] = wmma_bf16(aQ[kk], bK, S[nt]);
            }
        }

        // Online softmax stats per row (row r = i + 8*half, spread over 16 lanes)
        float alpha[8], mnew[8];
#pragma unroll
        for (int i = 0; i < 8; ++i) {
            float mx = -1e30f;
#pragma unroll
            for (int t = 0; t < 4; ++t) {
                S[t][i] *= scale;
                mx = fmaxf(mx, S[t][i]);
            }
            mx = fmaxf(mx, __shfl_xor(mx, 1));
            mx = fmaxf(mx, __shfl_xor(mx, 2));
            mx = fmaxf(mx, __shfl_xor(mx, 4));
            mx = fmaxf(mx, __shfl_xor(mx, 8));
            float mn = fmaxf(mrun[i], mx);
            alpha[i] = __expf(mrun[i] - mn);
            mrun[i]  = mn;
            mnew[i]  = mn;
            lrun[i] *= alpha[i];
        }

        // P = exp(S - m), R = leaky(S); via wave-private LDS for the
        // C-layout -> A-layout transpose; accumulate row sums, rescale accE.
#pragma unroll
        for (int i = 0; i < 8; ++i) {
            float ps = 0.f;
#pragma unroll
            for (int t = 0; t < 4; ++t) {
                float sv = S[t][i];
                float p  = __expf(sv - mnew[i]);
                float r  = (sv >= 0.f) ? sv : 0.01f * sv;
                pw[(i + 8 * half) * 64 + t * 16 + l16] = (bf16_t)p;
                rw[(i + 8 * half) * 64 + t * 16 + l16] = (bf16_t)r;
                ps += p;
            }
            ps += __shfl_xor(ps, 1);
            ps += __shfl_xor(ps, 2);
            ps += __shfl_xor(ps, 4);
            ps += __shfl_xor(ps, 8);
            lrun[i] += ps;
#pragma unroll
            for (int t = 0; t < 4; ++t) accE[t][i] *= alpha[i];
        }

        // Intra-wave LDS store->load hazard (cross-lane): drain DS counter.
        asm volatile("s_wait_dscnt 0" ::: "memory");

        v16bf pF[2], rF[2];
#pragma unroll
        for (int kk = 0; kk < 2; ++kk) {
            pF[kk] = load_afrag(pw + (size_t)l16 * 64 + kk * 32 + 8 * half);
            rF[kk] = load_afrag(rw + (size_t)l16 * 64 + kk * 32 + 8 * half);
        }

        // accE += P @ V ; accL += R @ V   (V tile from LDS, [d][s] local)
#pragma unroll
        for (int kk = 0; kk < 2; ++kk) {
#pragma unroll
            for (int dt = 0; dt < 4; ++dt) {
                v16bf bV = load_bfrag(&sV[buf][(dt * 16 + l16) * 72 + kk * 32 + 8 * half]);
                accE[dt] = wmma_bf16(pF[kk], bV, accE[dt]);
                accL[dt] = wmma_bf16(rF[kk], bV, accL[dt]);
            }
        }

        // Hand off the freshly DMA'd buffer: issuing wave drains TENSORcnt,
        // then the whole workgroup synchronizes.
        if (wid == 0) __builtin_amdgcn_s_wait_tensorcnt(0);
        __syncthreads();
    }

    // Epilogue: out = w1 * softmax-part + w2 * leaky-part  -> bf16 [b,s,e]
#pragma unroll
    for (int dt = 0; dt < 4; ++dt)
#pragma unroll
        for (int i = 0; i < 8; ++i) {
            float o = w1 * (accE[dt][i] / lrun[i]) + w2 * accL[dt][i];
            int s = qr0 + i + 8 * half;
            int e = h * HDIM + dt * 16 + l16;
            aout[((size_t)b * SEQ + s) * EMB + e] = (bf16_t)o;
        }
}

// ---------------------------------------------------------------------------
// Kernel 3: out = attn_out(bf16) @ Wout + bout  -> fp32 (M=4096, N=K=1024)
// A fragments direct from global (bf16 row-major == A layout), B staged in
// double-buffered LDS.
// ---------------------------------------------------------------------------
__global__ __launch_bounds__(256)
void aisa_proj_kernel(const bf16_t* __restrict__ A, const float* __restrict__ Wout,
                      const float* __restrict__ bout, float* __restrict__ out)
{
    __shared__ __align__(16) bf16_t sB[2][128 * 72];   // [n][k] transposed

    const int tid  = threadIdx.x;
    const int lane = tid & 31;
    const int wid  = tid >> 5;
    const int half = lane >> 4;
    const int l16  = lane & 15;
    const int n0   = blockIdx.x * 128;   // over 1024
    const int m0   = blockIdx.y * 128;   // over 4096
    const int waveM = wid >> 1;
    const int waveN = wid & 1;

    const v8f vz = {0.f,0.f,0.f,0.f,0.f,0.f,0.f,0.f};
    v8f acc[2][4];
#pragma unroll
    for (int mt = 0; mt < 2; ++mt)
#pragma unroll
        for (int nt = 0; nt < 4; ++nt) acc[mt][nt] = vz;

    float4 pb[8];
    auto fetch = [&](int k0) {
#pragma unroll
        for (int it = 0; it < 8; ++it) {
            int idx = tid + it * 256;
            pb[it] = *(const float4*)(Wout + (size_t)(k0 + (idx >> 5)) * EMB
                                      + n0 + ((idx & 31) << 2));
        }
    };
    auto stage = [&](int buf) {
#pragma unroll
        for (int it = 0; it < 8; ++it) {
            int idx = tid + it * 256;
            int rb = idx >> 5, cb4 = (idx & 31) << 2;
            sB[buf][(cb4 + 0) * 72 + rb] = (bf16_t)pb[it].x;
            sB[buf][(cb4 + 1) * 72 + rb] = (bf16_t)pb[it].y;
            sB[buf][(cb4 + 2) * 72 + rb] = (bf16_t)pb[it].z;
            sB[buf][(cb4 + 3) * 72 + rb] = (bf16_t)pb[it].w;
        }
    };

    fetch(0);
    stage(0);
    __syncthreads();

    for (int k0 = 0; k0 < EMB; k0 += 64) {
        const int buf = (k0 >> 6) & 1;
        const bool more = (k0 + 64) < EMB;
        if (more) fetch(k0 + 64);

#pragma unroll
        for (int kk = 0; kk < 64; kk += 32) {
            v16bf aF[2], bF[4];
#pragma unroll
            for (int mt = 0; mt < 2; ++mt)
                aF[mt] = load_afrag(A + (size_t)(m0 + waveM * 32 + mt * 16 + l16) * EMB
                                    + k0 + kk + 8 * half);
#pragma unroll
            for (int nt = 0; nt < 4; ++nt)
                bF[nt] = load_bfrag(&sB[buf][(waveN * 64 + nt * 16 + l16) * 72 + kk + 8 * half]);
#pragma unroll
            for (int mt = 0; mt < 2; ++mt)
#pragma unroll
                for (int nt = 0; nt < 4; ++nt)
                    acc[mt][nt] = wmma_bf16(aF[mt], bF[nt], acc[mt][nt]);
        }

        if (more) stage(buf ^ 1);
        __syncthreads();
    }

#pragma unroll
    for (int mt = 0; mt < 2; ++mt)
#pragma unroll
        for (int nt = 0; nt < 4; ++nt)
#pragma unroll
            for (int i = 0; i < 8; ++i) {
                int gr = m0 + waveM * 32 + mt * 16 + i + 8 * half;
                int gc = n0 + waveN * 64 + nt * 16 + l16;
                out[(size_t)gr * EMB + gc] = acc[mt][nt][i] + bout[gc];
            }
}

// ---------------------------------------------------------------------------
extern "C" void kernel_launch(void* const* d_in, const int* in_sizes, int n_in,
                              void* d_out, int out_size, void* d_ws, size_t ws_size,
                              hipStream_t stream) {
    (void)in_sizes; (void)n_in; (void)out_size; (void)ws_size;
    const float* x    = (const float*)d_in[0];
    const float* Wqkv = (const float*)d_in[1];
    const float* W1   = (const float*)d_in[2];
    const float* W2   = (const float*)d_in[3];
    const float* Wout = (const float*)d_in[4];
    const float* bout = (const float*)d_in[5];
    float* out = (float*)d_out;

    char* ws = (char*)d_ws;
    const size_t SEG = (size_t)8 * 1024 * 1024;     // 4M bf16 each
    bf16_t* qb = (bf16_t*)(ws + 0 * SEG);
    bf16_t* kb = (bf16_t*)(ws + 1 * SEG);
    bf16_t* vT = (bf16_t*)(ws + 2 * SEG);
    bf16_t* ao = (bf16_t*)(ws + 3 * SEG);

    aisa_qkv_kernel <<<dim3(3072 / 128, 4096 / 128), 256, 0, stream>>>(x, Wqkv, qb, kb, vT);
    aisa_attn_kernel<<<dim3(BATCH * NHEAD * (SEQ / 128)), 256, 0, stream>>>(qb, kb, vT, W1, W2, ao);
    aisa_proj_kernel<<<dim3(EMB / 128, 4096 / 128), 256, 0, stream>>>(ao, Wout, bout, out);
}
